// MultiHeadAttentionBlock_66640712565008
// MI455X (gfx1250) — compile-verified, matrix operand order checked by reading, not run on hardware
//
#include <hip/hip_runtime.h>
#include <hip/hip_bf16.h>
#include <stdint.h>

// ---------------------------------------------------------------------------
// MHA block for MI455X (gfx1250, wave32, WMMA).
//   B=2, S=2048, D=1024, H=16, DK=64.
// Pipeline:
//   1) f32 -> bf16 conversion (wq/bq pre-scaled by log2e/sqrt(DK)) + mask bit-pack
//   2) WMMA GEMM  Q/K/V = x @ W^T + b           (bf16 in, bf16 out)
//   3) flash-attention (online softmax in exp2 domain, bf16 WMMA, fp32 accum)
//   4) WMMA GEMM  out = ctx @ Wo^T + bo         (bf16 in, f32 out)
// ---------------------------------------------------------------------------

typedef __attribute__((ext_vector_type(16))) __bf16 v16bf;
typedef __attribute__((ext_vector_type(8)))  float  v8f;

union Frag16 { v16bf v; uint32_t u[8]; };

__device__ inline float fast_exp2(float x) {
    return __builtin_amdgcn_exp2f(x);          // bare v_exp_f32
}

__device__ inline v8f wmma_bf16(v16bf a, v16bf b, v8f c) {
    // D = A(16x32) * B(32x16) + C(16x16 f32)
    return __builtin_amdgcn_wmma_f32_16x16x32_bf16(
        /*neg_a=*/false, a, /*neg_b=*/false, b,
        /*c_mod=*/(short)0, c, /*reuse_a=*/false, /*reuse_b=*/false);
}

// A-matrix fragment (16x32 bf16, M x K), also valid as B fragment when the
// source is B^T row-major (rows = n, k contiguous).  ISA 7.12.2 layout:
//   lane: m = lane&15, half = lane>>4 selects K+0 / K+8 split
//   dword v: kbase = (v>=4 ? 16 : 0) + (v&3)*2, elements k, k+1
__device__ inline v16bf load_frag_kcontig(const __bf16* __restrict__ src, int ld, int lane) {
    const int m = lane & 15, half = lane >> 4;
    const __bf16* p = src + (size_t)m * ld + half * 8;
    Frag16 f;
#pragma unroll
    for (int v = 0; v < 8; ++v) {
        const int kb = ((v & 4) << 2) + (v & 3) * 2;   // 0,2,4,6,16,18,20,22
        f.u[v] = *(const uint32_t*)(p + kb);
    }
    return f.v;
}

// B-matrix fragment (32x16 bf16, K x N) from a [k][n] row-major source
// (n contiguous, k strided): element (k, n) = src[k*ld + n].
__device__ inline v16bf load_b_frag_ncontig(const __bf16* __restrict__ src, int ld, int lane) {
    const int n = lane & 15, half = lane >> 4;
    const __bf16* p = src + (size_t)(half * 8) * ld + n;
    v16bf b;
#pragma unroll
    for (int v = 0; v < 8; ++v) {
        const int kb = ((v & 4) << 2) + (v & 3) * 2;
        b[2 * v]     = p[(size_t)kb * ld];
        b[2 * v + 1] = p[(size_t)(kb + 1) * ld];
    }
    return b;
}

// ---------------------------------------------------------------------------
struct bf16x4 { __bf16 x, y, z, w; };

__global__ void __launch_bounds__(256, 1)
f32_to_bf16_kernel(const float4* __restrict__ src, bf16x4* __restrict__ dst,
                   size_t n4, float scale) {
    size_t i = (size_t)blockIdx.x * blockDim.x + threadIdx.x;
    if (i < n4) {
        const float4 f = src[i];
        bf16x4 o;
        o.x = (__bf16)(f.x * scale); o.y = (__bf16)(f.y * scale);
        o.z = (__bf16)(f.z * scale); o.w = (__bf16)(f.w * scale);
        dst[i] = o;
    }
}

// Bit-pack the [S,S] int mask: mb[sk * (S/32) + w] bit j = (mask[(w*32+j)*S + sk] != 0)
__global__ void __launch_bounds__(256, 1)
pack_mask_kernel(const int* __restrict__ mask, uint32_t* __restrict__ mb, int S) {
    const int idx = blockIdx.x * blockDim.x + threadIdx.x;   // over S * S/32
    const int words = S >> 5;
    if (idx >= S * words) return;
    const int sk = idx / words, w = idx % words;
    uint32_t bits = 0;
#pragma unroll 4
    for (int j = 0; j < 32; ++j)
        bits |= (mask[(size_t)(w * 32 + j) * S + sk] != 0 ? 1u : 0u) << j;
    mb[idx] = bits;
}

// ---------------------------------------------------------------------------
// Y[m,n] = sum_k X[m,k] * W[n,k] + bias[n]*bscale
// X: [M,K] bf16 row-major, W: [N,K] bf16 row-major (nn.Linear weight).
// Block: 256 threads = 8 waves; tile 128(M) x 64(N); X and W tiles staged in LDS.
template <typename OutT>
__global__ void __launch_bounds__(256, 1)
gemm_bias_kernel(const __bf16* __restrict__ X,
                 const __bf16* __restrict__ W,
                 const float* __restrict__ bias,
                 OutT* __restrict__ Y,
                 int M, int N, int K, float bscale) {
    constexpr int LDW = 34;                       // 32 + 2 pad (odd dword stride)
    __shared__ __bf16 wsh[64 * LDW];
    __shared__ __bf16 xsh[128 * LDW];

    const int tid = threadIdx.x, wave = tid >> 5, lane = tid & 31;
    const int n16 = lane & 15, half = lane >> 4;
    const int row0 = blockIdx.y * 128;
    const int col0 = blockIdx.x * 64;

    // per-thread cooperative-load coordinates
    const int wn = tid >> 4, wk = (tid & 15) * 2;          // 16 rows x 16 dword-pairs

    v8f acc[4] = {};

    for (int k0 = 0; k0 < K; k0 += 32) {
        // prefetch next k-step (gfx1250 global_prefetch_b8)
        if (k0 + 32 < K) {
            __builtin_prefetch(&W[(size_t)(col0 + wn) * K + k0 + 32 + wk], 0, 1);
            __builtin_prefetch(&X[(size_t)(row0 + wn) * K + k0 + 32 + wk], 0, 1);
        }
        // cooperative stage: W tile 64(n) x 32(k), X tile 128(m) x 32(k)
#pragma unroll
        for (int i = 0; i < 4; ++i) {
            const int n = i * 16 + wn;
            *(uint32_t*)&wsh[n * LDW + wk] =
                *(const uint32_t*)&W[(size_t)(col0 + n) * K + k0 + wk];
        }
#pragma unroll
        for (int i = 0; i < 8; ++i) {
            const int m = i * 16 + wn;
            *(uint32_t*)&xsh[m * LDW + wk] =
                *(const uint32_t*)&X[(size_t)(row0 + m) * K + k0 + wk];
        }
        __syncthreads();

        const v16bf a = load_frag_kcontig(xsh + wave * 16 * LDW, LDW, lane);
#pragma unroll
        for (int nt = 0; nt < 4; ++nt) {
            const v16bf b = load_frag_kcontig(wsh + nt * 16 * LDW, LDW, lane);
            acc[nt] = wmma_bf16(a, b, acc[nt]);
        }
        __syncthreads();
    }

#pragma unroll
    for (int nt = 0; nt < 4; ++nt) {
        const int col = col0 + nt * 16 + n16;
        const float bv = bias[col] * bscale;
#pragma unroll
        for (int r = 0; r < 8; ++r) {
            const int row = row0 + wave * 16 + r + 8 * half;
            Y[(size_t)row * N + col] = (OutT)(acc[nt][r] + bv);
        }
    }
}

// ---------------------------------------------------------------------------
// Flash attention over one (b, h, 64-query tile) per block.
// 128 threads = 4 waves; wave w owns query rows [w*16, w*16+16).
// Q was pre-scaled by log2e/sqrt(DK): scores are already in the exp2 domain,
// so softmax uses bare v_exp_f32 (exp2) with no extra multiplies.
__global__ void __launch_bounds__(128, 1)
flash_attn_kernel(const __bf16* __restrict__ Qa,
                  const __bf16* __restrict__ Ka,
                  const __bf16* __restrict__ Va,
                  const uint32_t* __restrict__ mb,   // bit-packed mask [S][S/32]
                  __bf16* __restrict__ Xc) {
    constexpr int S = 2048, D = 1024, LDT = 72;   // 64 + 8 pad
    constexpr int MW = S >> 5;                    // mask words per key column
    // masked score = -1e-9 in the exp domain -> scale by log2e for exp2 domain
    constexpr float MASKVAL = -1.4426950408889634e-9f;
    __shared__ __bf16 ksh[64 * LDT];
    __shared__ __bf16 vsh[64 * LDT];
    __shared__ __bf16 psh[64 * LDT];

    const int tid = threadIdx.x, wave = tid >> 5, lane = tid & 31;
    const int n16 = lane & 15, half = lane >> 4;
    const int b = blockIdx.z, h = blockIdx.y, qt = blockIdx.x;
    const size_t rowbase = (size_t)b * S;
    const int col0 = h * 64;
    const int q0 = qt * 64 + wave * 16;           // sequence row of this wave
    const int qword = q0 >> 5, qbit = (q0 & 31) + 8 * half;

    // Q fragments for this wave's 16 rows (held for the whole K loop)
    const v16bf aq0 = load_frag_kcontig(Qa + (rowbase + q0) * D + col0,      D, lane);
    const v16bf aq1 = load_frag_kcontig(Qa + (rowbase + q0) * D + col0 + 32, D, lane);

    v8f o[4] = {};
    float mrow[8], lrow[8];
#pragma unroll
    for (int r = 0; r < 8; ++r) { mrow[r] = -3.0e38f; lrow[r] = 0.0f; }

    for (int kt = 0; kt < S / 64; ++kt) {
        // cooperative stage of K/V tiles: 64 keys x 64 dk
#pragma unroll
        for (int i = 0; i < 16; ++i) {
            const int p   = i * 128 + tid;
            const int row = p >> 5;
            const int k   = (p & 31) * 2;
            const size_t g = (rowbase + kt * 64 + row) * (size_t)D + col0 + k;
            *(uint32_t*)&ksh[row * LDT + k] = *(const uint32_t*)&Ka[g];
            *(uint32_t*)&vsh[row * LDT + k] = *(const uint32_t*)&Va[g];
        }
        // mask bits for this wave's 16 rows x 4 key subtiles (latency hidden
        // under the K/V staging); lane's 8 rows = 8 bits starting at qbit
        uint32_t tb[4];
#pragma unroll
        for (int nt = 0; nt < 4; ++nt)
            tb[nt] = (mb[(size_t)(kt * 64 + nt * 16 + n16) * MW + qword] >> qbit) & 0xffu;
        // prefetch next tile (gfx1250 global_prefetch_b8)
        if (kt + 1 < S / 64) {
            const size_t gn = (rowbase + (kt + 1) * 64 + (tid >> 1)) * (size_t)D + col0
                              + (tid & 1) * 32;
            __builtin_prefetch(&Ka[gn], 0, 1);
            __builtin_prefetch(&Va[gn], 0, 1);
        }
        __syncthreads();

        // scores: 16 q-rows x 64 keys  (Q pre-scaled => S = log2e * Q K^T / 8)
        v8f sf[4];
#pragma unroll
        for (int nt = 0; nt < 4; ++nt) {
            const v16bf bk0 = load_frag_kcontig(ksh + nt * 16 * LDT,      LDT, lane);
            const v16bf bk1 = load_frag_kcontig(ksh + nt * 16 * LDT + 32, LDT, lane);
            v8f s = {};
            s = wmma_bf16(aq0, bk0, s);
            s = wmma_bf16(aq1, bk1, s);
            // faithful to reference: masked positions get -1e-9 (exp2-scaled);
            // all-ones fast path: EXEC drops to 0, branch skipped
            if (__builtin_expect(tb[nt] != 0xffu, 0)) {
#pragma unroll
                for (int r = 0; r < 8; ++r)
                    if (((tb[nt] >> r) & 1u) == 0u) s[r] = MASKVAL;
            }
            sf[nt] = s;
        }

        // online softmax (exp2 domain), row-wise in 16-lane halves of the wave
#pragma unroll
        for (int r = 0; r < 8; ++r) {
            float mloc = fmaxf(fmaxf(sf[0][r], sf[1][r]), fmaxf(sf[2][r], sf[3][r]));
#pragma unroll
            for (int d = 8; d >= 1; d >>= 1)
                mloc = fmaxf(mloc, __shfl_xor(mloc, d, 32));
            const float mnew = fmaxf(mrow[r], mloc);
            const float corr = fast_exp2(mrow[r] - mnew);
            mrow[r] = mnew;

            float lsum = 0.0f;
#pragma unroll
            for (int nt = 0; nt < 4; ++nt) {
                const float pv = fast_exp2(sf[nt][r] - mnew);
                lsum += pv;
                psh[(wave * 16 + r + 8 * half) * LDT + nt * 16 + n16] = (__bf16)pv;
            }
#pragma unroll
            for (int d = 8; d >= 1; d >>= 1)
                lsum += __shfl_xor(lsum, d, 32);
            lrow[r] = lrow[r] * corr + lsum;
#pragma unroll
            for (int dt = 0; dt < 4; ++dt) o[dt][r] *= corr;
        }

        // O += P @ V   (reload P from LDS in A-fragment layout)
        const v16bf pa0 = load_frag_kcontig(psh + wave * 16 * LDT,      LDT, lane);
        const v16bf pa1 = load_frag_kcontig(psh + wave * 16 * LDT + 32, LDT, lane);
#pragma unroll
        for (int dt = 0; dt < 4; ++dt) {
            const v16bf bv0 = load_b_frag_ncontig(vsh + dt * 16,            LDT, lane);
            const v16bf bv1 = load_b_frag_ncontig(vsh + 32 * LDT + dt * 16, LDT, lane);
            o[dt] = wmma_bf16(pa0, bv0, o[dt]);
            o[dt] = wmma_bf16(pa1, bv1, o[dt]);
        }
        __syncthreads();
    }

    // normalize and store context (bf16, [B*S, D] with head-major columns)
#pragma unroll
    for (int dt = 0; dt < 4; ++dt) {
#pragma unroll
        for (int r = 0; r < 8; ++r) {
            const float val = o[dt][r] / lrow[r];
            const size_t row = rowbase + q0 + r + 8 * half;
            Xc[row * D + col0 + dt * 16 + n16] = (__bf16)val;
        }
    }
}

// ---------------------------------------------------------------------------
extern "C" void kernel_launch(void* const* d_in, const int* in_sizes, int n_in,
                              void* d_out, int out_size, void* d_ws, size_t ws_size,
                              hipStream_t stream) {
    const float* q   = (const float*)d_in[0];
    const float* kk  = (const float*)d_in[1];
    const float* vv  = (const float*)d_in[2];
    const int*   msk = (const int*)d_in[3];
    const float* wq  = (const float*)d_in[4];
    const float* bq  = (const float*)d_in[5];
    const float* wk  = (const float*)d_in[6];
    const float* bk  = (const float*)d_in[7];
    const float* wv  = (const float*)d_in[8];
    const float* bv  = (const float*)d_in[9];
    const float* wo  = (const float*)d_in[10];
    const float* bo  = (const float*)d_in[11];

    constexpr int    B  = 2, S = 2048, D = 1024, H = 16;
    constexpr size_t MD = (size_t)B * S * D;   // 4 Mi elems
    constexpr size_t WD = (size_t)D * D;       // 1 Mi elems
    // log2e / sqrt(DK): scores come out of WMMA ready for exp2-domain softmax
    constexpr float  RSCALE = 1.4426950408889634f * 0.125f;

    __bf16* p   = (__bf16*)d_ws;
    __bf16* qb  = p; p += MD;
    __bf16* kb_ = p; p += MD;
    __bf16* vb_ = p; p += MD;
    __bf16* wqb = p; p += WD;
    __bf16* wkb = p; p += WD;
    __bf16* wvb = p; p += WD;
    __bf16* wob = p; p += WD;
    __bf16* Qa  = p; p += MD;
    __bf16* Ka  = p; p += MD;
    __bf16* Va  = p; p += MD;
    __bf16* Xc  = p; p += MD;
    uint32_t* mb = (uint32_t*)p;               // S * S/32 words (512 KB)

    // 1) precision conversion (vectorized) + mask bit-pack
    auto conv = [&](const float* s, __bf16* d, size_t n, float sc) {
        const size_t n4 = n / 4;
        f32_to_bf16_kernel<<<dim3((unsigned)((n4 + 255) / 256)), 256, 0, stream>>>(
            (const float4*)s, (bf16x4*)d, n4, sc);
    };
    conv(q,  qb,  MD, 1.0f);
    conv(kk, kb_, MD, 1.0f);
    conv(vv, vb_, MD, 1.0f);
    conv(wq, wqb, WD, RSCALE);                 // fold log2e/sqrt(DK) into Q
    conv(wk, wkb, WD, 1.0f);
    conv(wv, wvb, WD, 1.0f);
    conv(wo, wob, WD, 1.0f);
    {
        const int nwords = S * (S / 32);
        pack_mask_kernel<<<dim3((nwords + 255) / 256), 256, 0, stream>>>(msk, mb, S);
    }

    // 2) QKV projections (Q bias also pre-scaled)
    dim3 ggrid(D / 64, (B * S) / 128);   // (16, 32)
    gemm_bias_kernel<__bf16><<<ggrid, 256, 0, stream>>>(qb,  wqb, bq, Qa, B * S, D, D, RSCALE);
    gemm_bias_kernel<__bf16><<<ggrid, 256, 0, stream>>>(kb_, wkb, bk, Ka, B * S, D, D, 1.0f);
    gemm_bias_kernel<__bf16><<<ggrid, 256, 0, stream>>>(vb_, wvb, bv, Va, B * S, D, D, 1.0f);

    // 3) flash attention -> context
    flash_attn_kernel<<<dim3(S / 64, H, B), 128, 0, stream>>>(Qa, Ka, Va, mb, Xc);

    // 4) output projection -> f32 d_out
    gemm_bias_kernel<float><<<ggrid, 256, 0, stream>>>(Xc, wob, bo, (float*)d_out, B * S, D, D, 1.0f);
}
